// DGN7_70428873720436
// MI455X (gfx1250) — matrix operand order repaired and checked by reading, not compiled, not run on hardware
//
#include <hip/hip_runtime.h>
#include <math.h>

// ---- problem constants (match reference setup) ----
#define BATCH   2
#define T_SEQ   4096
#define D_MODEL 1024
#define DH      32
#define KTOP    8
#define WAVES   8          // 256-thread block = 8 wave32s

typedef __attribute__((ext_vector_type(16))) __bf16 v16bf;
typedef __attribute__((ext_vector_type(8)))  __bf16 v8bf;
typedef __attribute__((ext_vector_type(8)))  float  v8f;
typedef __attribute__((ext_vector_type(4)))  unsigned int u32x4;
typedef __attribute__((ext_vector_type(4)))  int i32x4;
typedef __attribute__((ext_vector_type(8)))  int i32x8;

#if __has_include(<hip/amd_detail/amd_gfx1250_TDM.h>)
#define TDM_SIX_ARGS 1
#else
#define TDM_SIX_ARGS 0
#endif

// ---------------------------------------------------------------------------
// fp32 -> bf16 (RNE) conversion of x
// ---------------------------------------------------------------------------
__global__ void cvt_bf16_kernel(const float* __restrict__ x,
                                unsigned short* __restrict__ xh, int n) {
    int i = blockIdx.x * blockDim.x + threadIdx.x;
    int stride = gridDim.x * blockDim.x;
    for (; i < n; i += stride) {
        unsigned u = __float_as_uint(x[i]);
        unsigned r = u + 0x7FFFu + ((u >> 16) & 1u);
        xh[i] = (unsigned short)(r >> 16);
    }
}

// ---------------------------------------------------------------------------
// per-batch column mean of x  (reproduces the reference t=0 edge case:
// fully-masked softmax row -> uniform 1/T -> msg = mean over all positions)
// ---------------------------------------------------------------------------
__global__ void colmean_kernel(const float* __restrict__ x,
                               float* __restrict__ xmean) {
    int i = blockIdx.x * blockDim.x + threadIdx.x;   // 0 .. B*D-1
    int b = i / D_MODEL, d = i % D_MODEL;
    const float* p = x + (size_t)b * T_SEQ * D_MODEL + d;
    float s = 0.f;
    for (int t = 0; t < T_SEQ; ++t) s += p[(size_t)t * D_MODEL];
    xmean[i] = s * (1.0f / (float)T_SEQ);
}

// ---------------------------------------------------------------------------
// q,k projections: one wave per row; lane == head dim (DH == 32 == wave32)
// ---------------------------------------------------------------------------
__global__ __launch_bounds__(256) void qk_kernel(const float* __restrict__ x,
                                                 const float* __restrict__ Wq,
                                                 const float* __restrict__ Wk,
                                                 float* __restrict__ q,
                                                 float* __restrict__ k) {
    int gw   = (blockIdx.x * blockDim.x + threadIdx.x) >> 5;   // global wave = row
    int lane = threadIdx.x & 31;                               // head dim
    if (gw >= BATCH * T_SEQ) return;
    const float4* xv  = (const float4*)(x  + (size_t)gw  * D_MODEL);
    const float4* wqr = (const float4*)(Wq + (size_t)lane * D_MODEL);
    const float4* wkr = (const float4*)(Wk + (size_t)lane * D_MODEL);
    float aq = 0.f, ak = 0.f;
#pragma unroll 4
    for (int i = 0; i < D_MODEL / 4; ++i) {
        float4 xc = xv[i], a = wqr[i], b = wkr[i];
        aq += xc.x * a.x + xc.y * a.y + xc.z * a.z + xc.w * a.w;
        ak += xc.x * b.x + xc.y * b.y + xc.z * b.z + xc.w * b.w;
    }
    q[(size_t)gw * DH + lane] = aq;
    k[(size_t)gw * DH + lane] = ak;
}

// sorted (descending) top-8 insert; fully unrolled static-index exchange chain
__device__ __forceinline__ void topk_insert(float (&tv)[KTOP], int (&ti)[KTOP],
                                            float v, int j) {
    if (v > tv[KTOP - 1]) {
        tv[KTOP - 1] = v; ti[KTOP - 1] = j;
#pragma unroll
        for (int s = KTOP - 1; s > 0; --s) {
            bool  sw = tv[s] > tv[s - 1];
            float av = sw ? tv[s]     : tv[s - 1];
            float bv = sw ? tv[s - 1] : tv[s];
            int   ai = sw ? ti[s]     : ti[s - 1];
            int   bi = sw ? ti[s - 1] : ti[s];
            tv[s - 1] = av; tv[s] = bv;
            ti[s - 1] = ai; ti[s] = bi;
        }
    }
}

// ---------------------------------------------------------------------------
// main kernel: one block per 16-row strip.
//  stage 0: TDM (tensor_load_to_lds) stages the 16x1024 bf16 A strip
//  phase 1: WMMA sim tiles (bf16 in, f32 acc) + per-wave causal top-8
//  phase 2: 8-way merge, scores, softmax  (wave 0)
//  phase 3: msg gather + gelu epilogue    (all 256 threads)
// ---------------------------------------------------------------------------
__global__ __launch_bounds__(256) void topk_attn_kernel(
    const float* __restrict__ x, const unsigned short* __restrict__ xh,
    const float* __restrict__ q, const float* __restrict__ k,
    const float* __restrict__ gain, const float* __restrict__ bias,
    const float* __restrict__ log_mix, const float* __restrict__ log_scale,
    const float* __restrict__ xmean, float* __restrict__ out) {

    __shared__ __align__(16) unsigned short ldsA[16 * D_MODEL];  // 32 KB row strip (bf16)
    __shared__ float tileBuf[WAVES][16][17];                     // per-wave sim tile
    __shared__ float candV[16][WAVES * KTOP];
    __shared__ int   candI[16][WAVES * KTOP];
    __shared__ float wV[16][KTOP];
    __shared__ int   wI[16][KTOP];
    __shared__ int   wCnt[16];

    const int blk  = blockIdx.x;            // 0 .. B*(T/16)-1
    const int b    = blk / (T_SEQ / 16);
    const int it   = blk % (T_SEQ / 16);
    const int tid  = threadIdx.x;
    const int wave = tid >> 5;
    const int lane = tid & 31;

    const unsigned short* stripSrc =
        xh + ((size_t)b * T_SEQ + (size_t)it * 16) * D_MODEL;

    // ---- stage 0: move 16x1024 bf16 strip into LDS ----
#if __has_builtin(__builtin_amdgcn_tensor_load_to_lds)
    if (wave == 0) {
        // D# per CDNA5 ISA (2-D tile, groups 2/3 unused)
        unsigned long long ga = (unsigned long long)(uintptr_t)stripSrc;
        unsigned ldsOff = (unsigned)(uintptr_t)(&ldsA[0]);   // low 32 bits = LDS byte offset
        u32x4 g0;
        g0[0] = 1u;                                  // count=1, is_restore=0, no gather
        g0[1] = ldsOff;                              // lds_addr
        g0[2] = (unsigned)(ga & 0xFFFFFFFFu);        // global_addr[31:0]
        g0[3] = (unsigned)((ga >> 32) & 0x01FFFFFFu) // global_addr[56:32]
                | (2u << 30);                        // type = 2 ("image")
        i32x8 g1;
        g1[0] = (1 << 16);                           // wg_mask=0, data_size=1 (2 bytes)
        g1[1] = (int)(D_MODEL << 16);                // tensor_dim0[15:0]=1024 (<<16)
        g1[2] = (int)((D_MODEL >> 16) | (16u << 16));// tensor_dim0 hi | tensor_dim1=16
        g1[3] = (int)(D_MODEL << 16);                // tensor_dim1 hi=0 | tile_dim0=1024
        g1[4] = 16;                                  // tile_dim1=16, tile_dim2=0
        g1[5] = D_MODEL;                             // tensor_dim0_stride[31:0]=1024
        g1[6] = 0;                                   // stride0 hi | stride1 lo
        g1[7] = 0;                                   // stride1 hi
        i32x4 gz = {0, 0, 0, 0};
#if TDM_SIX_ARGS
        i32x8 gz8 = {0, 0, 0, 0, 0, 0, 0, 0};
        __builtin_amdgcn_tensor_load_to_lds(g0, g1, gz, gz, gz8, 0);
#else
        __builtin_amdgcn_tensor_load_to_lds(g0, g1, gz, gz, 0);
#endif
#if __has_builtin(__builtin_amdgcn_s_wait_tensorcnt)
        __builtin_amdgcn_s_wait_tensorcnt(0);
#else
        asm volatile("s_wait_tensorcnt 0x0" ::: "memory");
#endif
    }
#else
    {   // fallback: cooperative synchronous copy
        const uint4* src = (const uint4*)stripSrc;
        uint4* dst = (uint4*)ldsA;
        for (int i = tid; i < (16 * D_MODEL * 2) / 16; i += 256) dst[i] = src[i];
    }
#endif
    __syncthreads();

    float tv[KTOP]; int ti[KTOP];
#pragma unroll
    for (int s = 0; s < KTOP; ++s) { tv[s] = -INFINITY; ti[s] = -1; }

    const int rA = lane & 15;       // A row within strip (both half-waves)
    const int hi = lane >> 4;       // half-wave select

    // ---- phase 1: causal column tiles, interleaved across waves ----
    for (int jt = wave; jt <= it; jt += WAVES) {
        // prefetch next B tile for this wave (L2-resident x stays hot)
        if (jt + WAVES <= it) {
            const unsigned short* pf =
                xh + ((size_t)b * T_SEQ + (size_t)(jt + WAVES) * 16 + (lane & 15)) * D_MODEL;
            __builtin_prefetch(pf, 0, 1);
        }

        v8f acc = {};
        const unsigned short* bptr =
            xh + ((size_t)b * T_SEQ + (size_t)jt * 16 + (lane & 15)) * D_MODEL;

#pragma unroll 4
        for (int kk = 0; kk < D_MODEL; kk += 32) {
            // A (16x32 bf16): lane<16 -> K {0..7,16..23}; lane>=16 -> K {8..15,24..31}
            int ka = kk + hi * 8;
            v8bf a0 = *(const v8bf*)&ldsA[rA * D_MODEL + ka];
            v8bf a1 = *(const v8bf*)&ldsA[rA * D_MODEL + ka + 16];
            v16bf av = __builtin_shufflevector(a0, a1, 0, 1, 2, 3, 4, 5, 6, 7,
                                               8, 9, 10, 11, 12, 13, 14, 15);
            // B (32x16 bf16): lane<16 -> K {0..7,16..23}; lane>=16 -> K {8..15,24..31}
            int kb = kk + hi * 8;
            v8bf b0 = *(const v8bf*)(bptr + kb);
            v8bf b1 = *(const v8bf*)(bptr + kb + 16);
            v16bf bv = __builtin_shufflevector(b0, b1, 0, 1, 2, 3, 4, 5, 6, 7,
                                               8, 9, 10, 11, 12, 13, 14, 15);
            acc = __builtin_amdgcn_wmma_f32_16x16x32_bf16(
                false, av, false, bv, (short)0, acc, false, false);
        }

        // spill 16x16 tile (C layout: lane = N; VGPR r = M=r / M=8+r)
        int rc0 = (lane < 16) ? 0 : 8;
#pragma unroll
        for (int r = 0; r < 8; ++r)
            tileBuf[wave][rc0 + r][lane & 15] = acc[r] * (1.0f / 32.0f);  // /sqrt(D)

        // lanes 0..15: row-owner scan with causal mask; insert chain only runs
        // when the tile's masked row-max beats the current 8th-best.
        if (lane < 16) {
            int tGlob = it * 16 + lane;
            float vals[16];
            float mx = -INFINITY;
#pragma unroll
            for (int c = 0; c < 16; ++c) {
                int j = jt * 16 + c;
                float v = tileBuf[wave][lane][c];
                vals[c] = (j < tGlob) ? v : -INFINITY;
                mx = fmaxf(mx, vals[c]);
            }
            if (mx > tv[KTOP - 1]) {
#pragma unroll
                for (int c = 0; c < 16; ++c)
                    topk_insert(tv, ti, vals[c], jt * 16 + c);
            }
        }
    }

    if (lane < 16) {
#pragma unroll
        for (int s = 0; s < KTOP; ++s) {
            candV[lane][wave * KTOP + s] = tv[s];
            candI[lane][wave * KTOP + s] = ti[s];
        }
    }
    __syncthreads();

    // ---- phase 2: merge 8 partial lists, score + softmax (wave 0) ----
    if (wave == 0 && lane < 16) {
#pragma unroll
        for (int s = 0; s < KTOP; ++s) { tv[s] = -INFINITY; ti[s] = -1; }
        for (int u = 0; u < WAVES * KTOP; ++u) {
            float v = candV[lane][u];
            int   j = candI[lane][u];
            if (j >= 0) topk_insert(tv, ti, v, j);
        }

        int t = it * 16 + lane;
        const float* qr = q + ((size_t)b * T_SEQ + t) * DH;
        float sc[KTOP];
        int cnt = 0;
        float mx = -INFINITY;
#pragma unroll
        for (int s = 0; s < KTOP; ++s) {
            float v = -INFINITY;
            int j = ti[s];
            if (j >= 0) {
                const float* kr = k + ((size_t)b * T_SEQ + j) * DH;
                float d = 0.f;
#pragma unroll
                for (int h = 0; h < DH; h += 4)
                    d += qr[h] * kr[h] + qr[h + 1] * kr[h + 1] +
                         qr[h + 2] * kr[h + 2] + qr[h + 3] * kr[h + 3];
                v = d * 0.17677669529663687f;  // 1/sqrt(DH)
                cnt++;
                mx = fmaxf(mx, v);
            }
            sc[s] = v;
        }
        float sum = 0.f;
#pragma unroll
        for (int s = 0; s < KTOP; ++s) {
            float e = (ti[s] >= 0) ? expf(sc[s] - mx) : 0.f;
            sc[s] = e; sum += e;
        }
        float inv = (cnt > 0) ? 1.f / sum : 0.f;
#pragma unroll
        for (int s = 0; s < KTOP; ++s) {
            wV[lane][s] = sc[s] * inv;
            wI[lane][s] = (ti[s] >= 0) ? ti[s] : 0;
        }
        wCnt[lane] = cnt;
    }
    __syncthreads();

    // ---- phase 3: msg gather + blend + gelu epilogue (all 256 threads) ----
    float lm = log_mix[0], ls = log_scale[0];
    float mix   = 1.f / (1.f + expf(-lm));
    float scale = log1pf(expf(ls)) + 0.01f;

    int r = tid >> 4;        // row 0..15
    int c = tid & 15;        // 16 threads per row -> contiguous d per 16-group
    int t = it * 16 + r;
    size_t orow = ((size_t)b * T_SEQ + t) * D_MODEL;
    const float* xb = x + (size_t)b * T_SEQ * D_MODEL;

    int   cnt = wCnt[r];
    float w[KTOP]; int ix[KTOP];
#pragma unroll
    for (int s = 0; s < KTOP; ++s) { w[s] = wV[r][s]; ix[s] = wI[r][s]; }

    for (int d = c; d < D_MODEL; d += 16) {
        float msg;
        if (cnt == 0) {
            msg = xmean[b * D_MODEL + d];      // fully-masked row -> uniform softmax
        } else {
            msg = 0.f;
#pragma unroll
            for (int s = 0; s < KTOP; ++s)
                msg += w[s] * xb[(size_t)ix[s] * D_MODEL + d];
        }
        float xv = x[orow + d];
        float bl = mix * xv + (1.f - mix) * msg;
        float z  = bl * gain[d] + bias[d];
        float g  = 0.5f * z * (1.f + erff(z * 0.70710678118654752f));
        out[orow + d] = g * scale;
    }
}

// ---------------------------------------------------------------------------
extern "C" void kernel_launch(void* const* d_in, const int* in_sizes, int n_in,
                              void* d_out, int out_size, void* d_ws, size_t ws_size,
                              hipStream_t stream) {
    const float* x         = (const float*)d_in[0];
    const float* Wq        = (const float*)d_in[1];
    const float* Wk        = (const float*)d_in[2];
    const float* gain      = (const float*)d_in[3];
    const float* bias      = (const float*)d_in[4];
    const float* log_mix   = (const float*)d_in[5];
    const float* log_scale = (const float*)d_in[6];
    float* out = (float*)d_out;

    // workspace layout
    char* ws = (char*)d_ws;
    unsigned short* xh = (unsigned short*)ws;                        // 16 MB bf16 x
    float* q     = (float*)(ws + (size_t)BATCH * T_SEQ * D_MODEL * 2);
    float* k     = q + (size_t)BATCH * T_SEQ * DH;
    float* xmean = k + (size_t)BATCH * T_SEQ * DH;

    const int n = BATCH * T_SEQ * D_MODEL;

    cvt_bf16_kernel<<<4096, 256, 0, stream>>>(x, xh, n);
    colmean_kernel<<<(BATCH * D_MODEL) / 256, 256, 0, stream>>>(x, xmean);
    qk_kernel<<<(BATCH * T_SEQ) / 8, 256, 0, stream>>>(x, Wq, Wk, q, k);
    topk_attn_kernel<<<BATCH * (T_SEQ / 16), 256, 0, stream>>>(
        x, xh, q, k, gain, bias, log_mix, log_scale, xmean, out);
}
